// GRP_46926812677073
// MI455X (gfx1250) — matrix operand order repaired and checked by reading, not compiled, not run on hardware
//
#include <hip/hip_runtime.h>

typedef __attribute__((ext_vector_type(16))) _Float16 v16h;
typedef __attribute__((ext_vector_type(8)))  _Float16 v8h;
typedef __attribute__((ext_vector_type(8)))  float    v8f;

constexpr int Bb = 16, E = 768, NH = 12, HS = 64, LAY = 12;
constexpr int Tok = 621, Tp = 640, FF = 3072;
constexpr int NPAT = 588;                 // 196 goal-img + 392 img patches per batch
constexpr int MROWS = Bb * Tp;            // 10240 (80 * 128)
constexpr int PROWS = Bb * NPAT;          // 9408
constexpr int PROWS_PAD = 9472;           // 74 * 128
constexpr float SCL = 0.03608439182435161f;   // 1/sqrt(768)

union V16H { v16h v; v8h h[2]; };

// ---- mask of the reference _build_mask(), evaluated analytically ----
__device__ __forceinline__ bool keep_mask(int t, int s) {
  if (t == s) return true;
  if (s < 32)  return !(t >= 32 && t < 228);   // goal-img rows don't see text
  if (s < 228) return false;                   // goal-img cols: diagonal only
  if (s < 424) return t >= 228;                // image chunk 0
  if (s < 620) return t >= 424;                // image chunk 1
  return false;                                // action col / padding: diagonal only
}

__device__ __forceinline__ float wred_max16(float x) {
#pragma unroll
  for (int off = 8; off; off >>= 1) x = fmaxf(x, __shfl_xor(x, off));
  return x;
}
__device__ __forceinline__ float wred_sum16(float x) {
#pragma unroll
  for (int off = 8; off; off >>= 1) x += __shfl_xor(x, off);
  return x;
}

// ======================= generic WMMA GEMM with async-LDS B staging ==========
// C(M,N) = A(M,K) * Bt(N,K)^T.
// Workgroup = 8 waves = one 128(M) x 64(N) macro-tile; grid sized exactly.
// B tile (64 cols x 32 k, shared by all 8 waves) is staged in LDS with
// GLOBAL_LOAD_ASYNC_TO_LDS_B128 (ASYNCcnt), double-buffered:
//   copy of k-step i+1 overlaps WMMAs of k-step i; s_wait_asynccnt 1 gates reuse.
// EPI: 0 = patch-embed store into X (row -> token p+32, +bias; rows >= Mreal clipped)
//      1 = fused QKV scatter (Q,K row-major; V transposed), no bias
//      2 = f32 residual accumulate (+bias)
//      3 = f16 store (+bias, optional ReLU)
template <int EPI, bool RELU>
__global__ __launch_bounds__(256) void gemm_wmma(
    const _Float16* __restrict__ A, int lda,
    const _Float16* __restrict__ Bt, int ldb,
    const float* __restrict__ bias,
    void* __restrict__ outp, int ldc,
    int Mreal, int N, int Kd)
{
  // 64 rows x 32 halves, row pitch 40 halves (80B): bank-stride 20 -> conflict-free
  __shared__ __align__(16) _Float16 bst[2][64][40];
  int tid  = threadIdx.x;
  int wv   = tid >> 5;
  int lane = tid & 31;
  int tilesN = N >> 6;
  int m0 = (blockIdx.x / tilesN) * 128 + wv * 16;
  int n0 = (blockIdx.x % tilesN) * 64;
  int lm = lane & 15, g = lane >> 4;

  // async-copy mapping: 256 threads -> 64 rows x 4 chunks of 16B
  int crow = tid >> 2;              // 0..63
  int cchk = (tid & 3) * 8;         // half-element offset 0,8,16,24
  const _Float16* gsrc = Bt + (size_t)(n0 + crow) * ldb + cchk;

  v8f acc[4];
#pragma unroll
  for (int c = 0; c < 4; ++c)
#pragma unroll
    for (int i = 0; i < 8; ++i) acc[c][i] = 0.0f;

  const _Float16* arow = A + (size_t)(m0 + lm) * lda;
  int nk = Kd >> 5;

  {
    unsigned loff = (unsigned)(uintptr_t)(&bst[0][crow][cchk]);
    const _Float16* gp = gsrc;
    asm volatile("global_load_async_to_lds_b128 %0, %1, off"
                 :: "v"(loff), "v"(gp) : "memory");
  }
  for (int ks = 0; ks < nk; ++ks) {
    int kk = ks << 5;
    if (ks + 1 < nk) {                               // prefetch next k-step
      unsigned loff = (unsigned)(uintptr_t)(&bst[(ks + 1) & 1][crow][cchk]);
      const _Float16* gp = gsrc + kk + 32;
      asm volatile("global_load_async_to_lds_b128 %0, %1, off"
                   :: "v"(loff), "v"(gp) : "memory");
      asm volatile("s_wait_asynccnt 0x1" ::: "memory");
    } else {
      asm volatile("s_wait_asynccnt 0x0" ::: "memory");
    }
    __syncthreads();                                 // B tile visible to all waves

    V16H a;                                          // A layout: halves K g*16..g*16+15
    a.h[0] = *(const v8h*)(arow + kk + g * 16);
    a.h[1] = *(const v8h*)(arow + kk + g * 16 + 8);
    int buf = ks & 1;
#pragma unroll
    for (int c = 0; c < 4; ++c) {
      V16H b;                                        // B layout: halves K g*16..g*16+15
      b.h[0] = *(const v8h*)(&bst[buf][c * 16 + lm][g * 16]);
      b.h[1] = *(const v8h*)(&bst[buf][c * 16 + lm][g * 16 + 8]);
      acc[c] = __builtin_amdgcn_wmma_f32_16x16x32_f16(false, a.v, false, b.v,
                                                      (short)0, acc[c], false, false);
    }
    __syncthreads();                                 // done reading before overwrite
  }

#pragma unroll
  for (int c = 0; c < 4; ++c) {
    int n = n0 + c * 16 + lm;
    float bv = bias ? bias[n] : 0.0f;
#pragma unroll
    for (int r = 0; r < 8; ++r) {
      int row = m0 + g * 8 + r;                      // C layout: lane=col, vgpr=row
      if (EPI == 0 && row >= Mreal) continue;
      float val = acc[c][r] + bv;
      if (EPI == 0) {
        int b = row / NPAT, p = row % NPAT;          // token t = p + 32
        ((float*)outp)[((size_t)b * Tp + (p + 32)) * E + n] = val;
      } else if (EPI == 1) {
        int b = row / Tp, t = row % Tp;
        int which = n / E, cc = n % E, h = cc / HS, d = cc % HS;
        size_t qsz = (size_t)Bb * NH * Tp * HS;
        _Float16* base = (_Float16*)outp + (size_t)which * qsz;
        _Float16 hv = (_Float16)val;
        if (which < 2) base[(((size_t)b * NH + h) * Tp + t) * HS + d] = hv;  // Q,K
        else           base[(((size_t)b * NH + h) * HS + d) * Tp + t] = hv;  // V^T
      } else if (EPI == 2) {
        float* o = (float*)outp + (size_t)row * ldc + n;
        *o += val;
      } else {
        if (RELU) val = fmaxf(val, 0.0f);
        ((_Float16*)outp)[(size_t)row * ldc + n] = (_Float16)val;
      }
    }
  }
}

// ======================= flash attention =======================
// 8 waves/block; each wave owns one (b,h, 16-row) tile; online softmax over 32-wide s-blocks.
__global__ __launch_bounds__(256) void attn_kernel(
    const _Float16* __restrict__ Q, const _Float16* __restrict__ Kmat,
    const _Float16* __restrict__ Vt, _Float16* __restrict__ O)
{
  __shared__ __align__(16) _Float16 plds[8][16][32];
  int wib  = threadIdx.x >> 5;
  int lane = threadIdx.x & 31;
  int tile = (blockIdx.x << 3) + wib;
  int tb = tile % (Tp / 16);
  int bh = tile / (Tp / 16);
  int h = bh % NH, b = bh / NH;
  int t0 = tb << 4;
  int lm = lane & 15, g = lane >> 4;

  const _Float16* qrow = Q + ((size_t)bh * Tp + t0 + lm) * HS;
  V16H aq0, aq1;                                     // Q rows as A operand, K-dim 64
  aq0.h[0] = *(const v8h*)(qrow + g * 16);
  aq0.h[1] = *(const v8h*)(qrow + g * 16 + 8);
  aq1.h[0] = *(const v8h*)(qrow + 32 + g * 16);
  aq1.h[1] = *(const v8h*)(qrow + 32 + g * 16 + 8);

  float mrun[8], lrun[8];
  v8f oacc[4];
#pragma unroll
  for (int r = 0; r < 8; ++r) { mrun[r] = -1e30f; lrun[r] = 0.0f; }
#pragma unroll
  for (int c = 0; c < 4; ++c)
#pragma unroll
    for (int i = 0; i < 8; ++i) oacc[c][i] = 0.0f;

  for (int sb = 0; sb < Tp; sb += 32) {
    v8f c0, c1;
#pragma unroll
    for (int i = 0; i < 8; ++i) { c0[i] = 0.0f; c1[i] = 0.0f; }
    const _Float16* k0 = Kmat + ((size_t)bh * Tp + sb + lm) * HS;  // B cols = K rows
    const _Float16* k1 = k0 + 16 * HS;
    V16H b00, b01, b10, b11;
    b00.h[0] = *(const v8h*)(k0 + g * 16);      b00.h[1] = *(const v8h*)(k0 + g * 16 + 8);
    b01.h[0] = *(const v8h*)(k0 + 32 + g * 16); b01.h[1] = *(const v8h*)(k0 + 32 + g * 16 + 8);
    b10.h[0] = *(const v8h*)(k1 + g * 16);      b10.h[1] = *(const v8h*)(k1 + g * 16 + 8);
    b11.h[0] = *(const v8h*)(k1 + 32 + g * 16); b11.h[1] = *(const v8h*)(k1 + 32 + g * 16 + 8);
    c0 = __builtin_amdgcn_wmma_f32_16x16x32_f16(false, aq0.v, false, b00.v, (short)0, c0, false, false);
    c0 = __builtin_amdgcn_wmma_f32_16x16x32_f16(false, aq1.v, false, b01.v, (short)0, c0, false, false);
    c1 = __builtin_amdgcn_wmma_f32_16x16x32_f16(false, aq0.v, false, b10.v, (short)0, c1, false, false);
    c1 = __builtin_amdgcn_wmma_f32_16x16x32_f16(false, aq1.v, false, b11.v, (short)0, c1, false, false);

    int col0 = sb + lm, col1 = sb + 16 + lm;
    float p0[8], p1[8];
#pragma unroll
    for (int r = 0; r < 8; ++r) {
      int t = t0 + g * 8 + r;
      float s0 = keep_mask(t, col0) ? c0[r] * SCL : -1e30f;
      float s1 = keep_mask(t, col1) ? c1[r] * SCL : -1e30f;
      float mnew = fmaxf(mrun[r], wred_max16(fmaxf(s0, s1)));
      float alpha = __expf(mrun[r] - mnew);
      mrun[r] = mnew;
      float e0 = __expf(s0 - mnew), e1 = __expf(s1 - mnew);
      lrun[r] = lrun[r] * alpha + wred_sum16(e0 + e1);
      p0[r] = e0; p1[r] = e1;
#pragma unroll
      for (int c = 0; c < 4; ++c) oacc[c][r] *= alpha;
    }

    __syncthreads();                                  // transpose P via LDS
#pragma unroll
    for (int r = 0; r < 8; ++r) {
      plds[wib][g * 8 + r][lm]      = (_Float16)p0[r];
      plds[wib][g * 8 + r][16 + lm] = (_Float16)p1[r];
    }
    __syncthreads();
    V16H ap;
    ap.h[0] = *(const v8h*)(&plds[wib][lm][g * 16]);
    ap.h[1] = *(const v8h*)(&plds[wib][lm][g * 16 + 8]);
#pragma unroll
    for (int c = 0; c < 4; ++c) {
      const _Float16* vrow = Vt + ((size_t)bh * HS + c * 16 + lm) * Tp + sb + g * 16;
      V16H bv;
      bv.h[0] = *(const v8h*)(vrow);
      bv.h[1] = *(const v8h*)(vrow + 8);
      oacc[c] = __builtin_amdgcn_wmma_f32_16x16x32_f16(false, ap.v, false, bv.v,
                                                       (short)0, oacc[c], false, false);
    }
  }

#pragma unroll
  for (int c = 0; c < 4; ++c)
#pragma unroll
    for (int r = 0; r < 8; ++r) {
      int t = t0 + g * 8 + r;
      O[((size_t)b * Tp + t) * E + h * HS + c * 16 + lm] = (_Float16)(oacc[c][r] / lrun[r]);
    }
}

// ======================= LayerNorm: one wave per row =======================
__global__ __launch_bounds__(256) void ln_kernel(
    const float* __restrict__ X, const float* __restrict__ gain,
    const float* __restrict__ bias, _Float16* __restrict__ out, int rows)
{
  int wid = (blockIdx.x * blockDim.x + threadIdx.x) >> 5;
  int lane = threadIdx.x & 31;
  if (wid >= rows) return;
  const float* xr = X + (size_t)wid * E;
  float v[24]; float s = 0.0f;
#pragma unroll
  for (int i = 0; i < 24; ++i) { v[i] = xr[lane + 32 * i]; s += v[i]; }
#pragma unroll
  for (int off = 16; off; off >>= 1) s += __shfl_xor(s, off);
  float mean = s * (1.0f / 768.0f);
  float var = 0.0f;
#pragma unroll
  for (int i = 0; i < 24; ++i) { float d = v[i] - mean; var += d * d; }
#pragma unroll
  for (int off = 16; off; off >>= 1) var += __shfl_xor(var, off);
  float inv = rsqrtf(var * (1.0f / 768.0f) + 1e-5f);
  _Float16* orow = out + (size_t)wid * E;
#pragma unroll
  for (int i = 0; i < 24; ++i) {
    int e = lane + 32 * i;
    orow[e] = (_Float16)((v[i] - mean) * inv * gain[e] + bias[e]);
  }
}

// ======================= prep / assembly kernels =======================
__global__ void cvt_transpose(const float* __restrict__ in, _Float16* __restrict__ out,
                              int Kd, int Nd) {      // in (Kd,Nd) -> out (Nd,Kd)
  int idx = blockIdx.x * blockDim.x + threadIdx.x;
  if (idx >= Kd * Nd) return;
  int n = idx % Nd, k = idx / Nd;
  out[(size_t)n * Kd + k] = (_Float16)in[idx];
}

__global__ void cvt_qkv(const float* __restrict__ wq, const float* __restrict__ wk,
                        const float* __restrict__ wv, _Float16* __restrict__ out) {
  int idx = blockIdx.x * blockDim.x + threadIdx.x;   // out (2304, 768)
  if (idx >= 3 * E * E) return;
  int n = idx / E, e = idx % E;
  int w = n / E;                                     // 0=q,1=k,2=v
  int c = n % E, h = c / HS, d = c % HS;
  const float* src = (w == 0) ? wq : (w == 1) ? wk : wv;   // each (NH,E,HS)
  out[idx] = (_Float16)src[((size_t)h * E + e) * HS + d];
}

__global__ void patch_extract(const float* __restrict__ imgs,
                              const float* __restrict__ gimgs,
                              _Float16* __restrict__ Pbuf) {
  int idx = blockIdx.x * blockDim.x + threadIdx.x;   // PROWS * 768
  if (idx >= PROWS * E) return;
  int e = idx % E, row = idx / E;
  int b = row / NPAT, p = row % NPAT;
  int c3 = e % 3, pp = e / 3, ppx = pp % 16, ppy = pp / 16;
  const float* src;
  if (p < 196) {                                     // goal image (K=1)
    int py = p / 14, px = p % 14;
    src = gimgs + ((((size_t)b) * 224 + py * 16 + ppy) * 224 + px * 16 + ppx) * 3 + c3;
  } else {                                           // images (K=2)
    int pi = p - 196, kk = pi / 196, rem = pi % 196, py = rem / 14, px = rem % 14;
    src = imgs + ((((size_t)b * 2 + kk) * 224 + py * 16 + ppy) * 224 + px * 16 + ppx) * 3 + c3;
  }
  Pbuf[idx] = (_Float16)(*src);
}

__device__ __forceinline__ float pos_val(int t, int e) {
  float expo = (float)(e & ~1) / 768.0f;
  float ang = (float)t * __powf(10000.0f, -expo);
  return (e & 1) ? __cosf(ang) : __sinf(ang);
}

__global__ void assemble(float* __restrict__ X, const int* __restrict__ gtxt,
                         const float* __restrict__ temb, const float* __restrict__ aemb) {
  int idx = blockIdx.x * blockDim.x + threadIdx.x;   // MROWS * 768
  if (idx >= MROWS * E) return;
  int e = idx % E, t = (idx / E) % Tp, b = idx / (E * Tp);
  float* xp = X + idx;
  if (t >= Tok) { *xp = 0.0f; return; }
  float pos = pos_val(t, e);
  if (t < 32)        *xp = temb[(size_t)gtxt[b * 32 + t] * E + e] + pos;
  else if (t < 620)  *xp = *xp + pos;                // patch GEMM already stored embed+bias
  else               *xp = aemb[e] + pos;
}

// ======================= tiny head =======================
__global__ void head1(const float* __restrict__ X, const float* __restrict__ hw1,
                      const float* __restrict__ hb1, float* __restrict__ z1) {
  int idx = blockIdx.x * blockDim.x + threadIdx.x;
  if (idx >= Bb * E) return;
  int j = idx % E, b = idx / E;
  const float* xr = X + ((size_t)b * Tp + 620) * E;
  float s = hb1[j];
  for (int e = 0; e < E; ++e) s += xr[e] * hw1[(size_t)e * E + j];
  z1[idx] = fmaxf(s, 0.0f);
}
__global__ void head2(const float* __restrict__ z1, const float* __restrict__ hw2,
                      const float* __restrict__ hb2, float* __restrict__ out) {
  int idx = blockIdx.x * blockDim.x + threadIdx.x;
  if (idx >= Bb * 64) return;
  int a = idx % 64, b = idx / 64;
  float s = hb2[a];
  for (int j = 0; j < E; ++j) s += z1[b * E + j] * hw2[(size_t)j * 64 + a];
  out[idx] = s;
}

// ======================= launch =======================
extern "C" void kernel_launch(void* const* d_in, const int* in_sizes, int n_in,
                              void* d_out, int out_size, void* d_ws, size_t ws_size,
                              hipStream_t stream) {
  (void)in_sizes; (void)n_in; (void)out_size; (void)ws_size;
  const float* images     = (const float*)d_in[0];
  const int*   goals_txt  = (const int*)d_in[1];
  const float* goal_imgs  = (const float*)d_in[2];
  const float* patch_w    = (const float*)d_in[3];
  const float* patch_b    = (const float*)d_in[4];
  const float* text_emb   = (const float*)d_in[5];
  const float* action_emb = (const float*)d_in[6];
  const float* wq   = (const float*)d_in[7];
  const float* wk   = (const float*)d_in[8];
  const float* wv   = (const float*)d_in[9];
  const float* proj_w = (const float*)d_in[10];
  const float* proj_b = (const float*)d_in[11];
  const float* ln1_s  = (const float*)d_in[12];
  const float* ln1_b  = (const float*)d_in[13];
  const float* ln2_s  = (const float*)d_in[14];
  const float* ln2_b  = (const float*)d_in[15];
  const float* ff_w1  = (const float*)d_in[16];
  const float* ff_b1  = (const float*)d_in[17];
  const float* ff_w2  = (const float*)d_in[18];
  const float* ff_b2  = (const float*)d_in[19];
  const float* hw1 = (const float*)d_in[20];
  const float* hb1 = (const float*)d_in[21];
  const float* hw2 = (const float*)d_in[22];
  const float* hb2 = (const float*)d_in[23];

  size_t off = 0;
  auto carve = [&](size_t bytes) -> void* {
    void* r = (char*)d_ws + off;
    off += (bytes + 255) & ~(size_t)255;
    return r;
  };
  float*    X     = (float*)carve((size_t)MROWS * E * 4);
  _Float16* Pbuf  = (_Float16*)carve((size_t)PROWS_PAD * E * 2);
  _Float16* Hbuf  = (_Float16*)carve((size_t)MROWS * E * 2);
  size_t qsz = (size_t)Bb * NH * Tp * HS;
  _Float16* QKV   = (_Float16*)carve(3 * qsz * 2);
  _Float16* Obuf  = (_Float16*)carve((size_t)MROWS * E * 2);
  _Float16* H2    = (_Float16*)carve((size_t)MROWS * FF * 2);
  _Float16* pwT   = (_Float16*)carve((size_t)E * E * 2);
  _Float16* WqkvT = (_Float16*)carve((size_t)3 * E * E * 2);
  _Float16* projT = (_Float16*)carve((size_t)E * E * 2);
  _Float16* ff1T  = (_Float16*)carve((size_t)E * FF * 2);
  _Float16* ff2T  = (_Float16*)carve((size_t)E * FF * 2);
  float*    zbuf  = (float*)carve((size_t)Bb * E * 4);

  dim3 blk(256);
  auto g128 = [](int Mpad, int N) { return (Mpad / 128) * (N / 64); };

  cvt_transpose<<<(E * E + 255) / 256, blk, 0, stream>>>(patch_w, pwT, E, E);
  patch_extract<<<(PROWS * E + 255) / 256, blk, 0, stream>>>(images, goal_imgs, Pbuf);
  gemm_wmma<0, false><<<g128(PROWS_PAD, E), blk, 0, stream>>>(
      Pbuf, E, pwT, E, patch_b, X, E, PROWS, E, E);
  assemble<<<(MROWS * E + 255) / 256, blk, 0, stream>>>(X, goals_txt, text_emb, action_emb);

  for (int l = 0; l < LAY; ++l) {
    cvt_qkv<<<(3 * E * E + 255) / 256, blk, 0, stream>>>(
        wq + (size_t)l * NH * E * HS, wk + (size_t)l * NH * E * HS,
        wv + (size_t)l * NH * E * HS, WqkvT);
    cvt_transpose<<<(E * E + 255) / 256, blk, 0, stream>>>(proj_w + (size_t)l * E * E, projT, E, E);
    cvt_transpose<<<(E * FF + 255) / 256, blk, 0, stream>>>(ff_w1 + (size_t)l * E * FF, ff1T, E, FF);
    cvt_transpose<<<(E * FF + 255) / 256, blk, 0, stream>>>(ff_w2 + (size_t)l * FF * E, ff2T, FF, E);

    ln_kernel<<<MROWS / 8, blk, 0, stream>>>(X, ln1_s + (size_t)l * E, ln1_b + (size_t)l * E, Hbuf, MROWS);
    gemm_wmma<1, false><<<g128(MROWS, 3 * E), blk, 0, stream>>>(
        Hbuf, E, WqkvT, E, nullptr, QKV, 0, MROWS, 3 * E, E);
    attn_kernel<<<Bb * NH * (Tp / 16) / 8, blk, 0, stream>>>(QKV, QKV + qsz, QKV + 2 * qsz, Obuf);
    gemm_wmma<2, false><<<g128(MROWS, E), blk, 0, stream>>>(
        Obuf, E, projT, E, proj_b + (size_t)l * E, X, E, MROWS, E, E);
    ln_kernel<<<MROWS / 8, blk, 0, stream>>>(X, ln2_s + (size_t)l * E, ln2_b + (size_t)l * E, Hbuf, MROWS);
    gemm_wmma<3, true><<<g128(MROWS, FF), blk, 0, stream>>>(
        Hbuf, E, ff1T, E, ff_b1 + (size_t)l * FF, H2, FF, MROWS, FF, E);
    gemm_wmma<2, false><<<g128(MROWS, E), blk, 0, stream>>>(
        H2, FF, ff2T, FF, ff_b2 + (size_t)l * E, X, E, MROWS, E, FF);
  }

  head1<<<(Bb * E + 255) / 256, blk, 0, stream>>>(X, hw1, hb1, zbuf);
  head2<<<(Bb * 64 + 255) / 256, blk, 0, stream>>>(zbuf, hw2, hb2, (float*)d_out);
}
